// ScaledDotProductAttention_65618510348734
// MI455X (gfx1250) — matrix operand order repaired and checked by reading, not compiled, not run on hardware
//
#include <hip/hip_runtime.h>

constexpr int B_ = 2, H_ = 16, S_ = 2048, DK_ = 64;
constexpr float SCALE = 0.125f; // 1/sqrt(64)

typedef __attribute__((ext_vector_type(16))) __bf16 v16bf;
typedef __attribute__((ext_vector_type(2)))  __bf16 v2bf;
typedef __attribute__((ext_vector_type(8)))  float  v8f;

union BfVec { unsigned u[8]; v16bf v; };

// Native f32 -> bf16 conversions (lower to v_cvt_pk_bf16_f32-class ops).
__device__ __forceinline__ unsigned short f2bf(float f) {
    union { __bf16 b; unsigned short u; } r;
    r.b = (__bf16)f;
    return r.u;
}
__device__ __forceinline__ unsigned pack2(float lo, float hi) {
    union { v2bf v; unsigned u; } r;
    r.v.x = (__bf16)lo;
    r.v.y = (__bf16)hi;
    return r.u;
}

// One workgroup = 8 wave32 = 256 threads, handles 16 query rows of one (b,h).
// LDS: score strip 16x2048 f32 (128KB) + Q tile bf16 + reduction scratch.
__global__ void ScaledDotProductAttention_65618510348734_kernel(
    const float* __restrict__ Q, const float* __restrict__ K,
    const float* __restrict__ V, const int* __restrict__ mask,
    float* __restrict__ out, float* __restrict__ wts) {
    extern __shared__ float smem[];
    float* Ss = smem;                                        // 16*2048 f32
    unsigned short* Qs = (unsigned short*)(Ss + 16 * S_);    // 16*64 bf16
    float* red     = (float*)(Qs + 16 * DK_);                // 256 f32
    float* rowstat = red + 256;                              // 16 f32
    float* pvred   = rowstat + 16;                           // 4*256 f32

    const int qt = blockIdx.x, h = blockIdx.y, b = blockIdx.z;
    const int q0 = qt * 16;
    const size_t headOff = ((size_t)(b * H_ + h)) * S_ * DK_;
    const float* Qh = Q + headOff + (size_t)q0 * DK_;
    const float* Kh = K + headOff;
    const float* Vh = V + headOff;

    const int tid  = threadIdx.x;
    const int wave = tid >> 5, lane = tid & 31;
    const int m = lane & 15, hi = lane >> 4;

    // Stage Q tile in LDS as bf16 (coalesced).
    for (int i = tid; i < 16 * DK_; i += 256) Qs[i] = f2bf(Qh[i]);
    __syncthreads();

    // ---- Phase 1: scores = Q K^T (bf16 WMMA, f32 accum), raw into LDS ----
    {
        // A fragments (Q tile, 16x32 bf16 layout) are invariant over the key
        // loop: hoist both K-steps into registers once.
        BfVec aq[2];
#pragma unroll
        for (int s = 0; s < 2; ++s)
#pragma unroll
            for (int j = 0; j < 8; ++j) {
                int kp = ((j & 4) << 2) + hi * 8 + ((j & 3) << 1);
                aq[s].u[j] = *(const unsigned*)&Qs[m * DK_ + s * 32 + kp];
            }

        for (int t = wave; t < S_ / 16; t += 8) {
            const int kbase = t * 16;
            if (t + 8 < S_ / 16) // prefetch next K tile for this wave
                __builtin_prefetch(&Kh[(size_t)(kbase + 128 + m) * DK_], 0, 0);
            v8f c = {};
#pragma unroll
            for (int s = 0; s < 2; ++s) {
                BfVec bv;
#pragma unroll
                for (int j = 0; j < 8; ++j) {
                    // B (32x16 bf16): lane col = kbase+m; VGPR j holds K = (j&4)*4+hi*8+2*(j&3), +1
                    int kb = s * 32 + ((j & 4) << 2) + hi * 8 + ((j & 3) << 1);
                    const float* kp2 = &Kh[(size_t)(kbase + m) * DK_ + kb];
                    bv.u[j] = pack2(kp2[0], kp2[1]);
                }
                c = __builtin_amdgcn_wmma_f32_16x16x32_bf16(
                        false, aq[s].v, false, bv.v, (short)0, c, false, false);
            }
#pragma unroll
            for (int v = 0; v < 8; ++v)
                Ss[(v + hi * 8) * S_ + kbase + m] = c[v];
        }
    }
    __syncthreads();

    // ---- Phase 2: f32 softmax over each row (scale + mask + max + exp + sum) ----
    {
        const int r = tid >> 4, sub = tid & 15; // 16 threads per row
        float* row = Ss + r * S_;
        const int* mrow = mask + (size_t)(q0 + r) * S_;
        float mx = -3.0e38f;
        for (int i = 0; i < S_ / 16; ++i) {
            int k = sub + i * 16; // stride-16 => coalesced across the wave
            float s = row[k] * SCALE;
            if (mrow[k] == 0) s = -1.0e9f;
            row[k] = s;
            mx = fmaxf(mx, s);
        }
        red[tid] = mx;
        __syncthreads();
        if (sub == 0) {
            float v = red[r * 16];
            for (int j = 1; j < 16; ++j) v = fmaxf(v, red[r * 16 + j]);
            rowstat[r] = v;
        }
        __syncthreads();
        float rm = rowstat[r];
        float sum = 0.f;
        for (int i = 0; i < S_ / 16; ++i) {
            int k = sub + i * 16;
            float p = __expf(row[k] - rm);
            row[k] = p;
            sum += p;
        }
        red[tid] = sum;
        __syncthreads();
        if (sub == 0) {
            float v = 0.f;
            for (int j = 0; j < 16; ++j) v += red[r * 16 + j];
            rowstat[r] = 1.0f / v;
        }
        __syncthreads();
        float inv = rowstat[r];
        float* wrow = wts + ((size_t)((b * H_ + h) * S_ + q0 + r)) * S_;
        for (int i = 0; i < S_ / 16; ++i) {
            int k = sub + i * 16;
            float p = row[k] * inv;
            row[k] = p;      // keep normalized P in LDS for PV
            wrow[k] = p;     // materialize attention weights (mandatory output)
        }
    }
    __syncthreads();

    // ---- Phase 3: O = P V (bf16 WMMA over K=2048, split across 2 wave-halves) ----
    {
        const int dsub = wave & 3, half = wave >> 2;
        const int dcol = dsub * 16 + m;
        v8f c = {};
        const int kstart = half * (S_ / 2);
        for (int k0 = kstart; k0 < kstart + S_ / 2; k0 += 32) {
            if (k0 + 32 < kstart + S_ / 2) // prefetch upcoming V rows
                __builtin_prefetch(&Vh[(size_t)(k0 + 32 + hi * 16) * DK_ + dcol], 0, 0);
            BfVec a, bv;
#pragma unroll
            for (int j = 0; j < 8; ++j) {
                int kp = ((j & 4) << 2) + hi * 8 + ((j & 3) << 1);
                const float* pp = &Ss[m * S_ + k0 + kp];
                a.u[j] = pack2(pp[0], pp[1]);
                int kb = k0 + ((j & 4) << 2) + hi * 8 + ((j & 3) << 1);
                bv.u[j] = pack2(Vh[(size_t)kb * DK_ + dcol],
                                Vh[(size_t)(kb + 1) * DK_ + dcol]);
            }
            c = __builtin_amdgcn_wmma_f32_16x16x32_bf16(
                    false, a.v, false, bv.v, (short)0, c, false, false);
        }
        if (half == 1) {
#pragma unroll
            for (int v = 0; v < 8; ++v)
                pvred[dsub * 256 + (v + hi * 8) * 16 + m] = c[v];
        }
        __syncthreads();
        if (half == 0) {
#pragma unroll
            for (int v = 0; v < 8; ++v) {
                int row = v + hi * 8;
                float val = c[v] + pvred[dsub * 256 + row * 16 + m];
                out[headOff + (size_t)(q0 + row) * DK_ + dcol] = val;
            }
        }
    }
}

extern "C" void kernel_launch(void* const* d_in, const int* in_sizes, int n_in,
                              void* d_out, int out_size, void* d_ws, size_t ws_size,
                              hipStream_t stream) {
    const float* Q    = (const float*)d_in[0];
    const float* K    = (const float*)d_in[1];
    const float* V    = (const float*)d_in[2];
    const int*   mask = (const int*)d_in[3];
    float* out = (float*)d_out;
    float* wts = out + (size_t)B_ * H_ * S_ * DK_; // outputs concatenated: (output, weights)

    // LDS: 16*2048*4 + 16*64*2 + 256*4 + 16*4 + 1024*4 = 138304 bytes (< 320KB/WGP)
    const size_t smem_bytes = (size_t)16 * S_ * 4 + 16 * DK_ * 2 + 256 * 4 + 16 * 4 + 1024 * 4;
    (void)hipFuncSetAttribute(
        reinterpret_cast<const void*>(&ScaledDotProductAttention_65618510348734_kernel),
        hipFuncAttributeMaxDynamicSharedMemorySize, (int)smem_bytes);

    dim3 grid(S_ / 16, H_, B_); // 128 x 16 x 2 = 4096 workgroups
    dim3 block(256);            // 8 wave32
    ScaledDotProductAttention_65618510348734_kernel<<<grid, block, smem_bytes, stream>>>(
        Q, K, V, mask, out, wts);
}
